// CausalAlibiAttn_60842506715716
// MI455X (gfx1250) — matrix operand order repaired and checked by reading, not compiled, not run on hardware
//
#include <hip/hip_runtime.h>
#include <cstdint>

// Problem constants (match reference)
#define BB   2
#define SS   2048
#define DD   2048
#define HH   32
#define HKV  8
#define DH   64
#define WIN  1024
#define GROUP (HH / HKV)          // 4
#define KV_D  (HKV * DH)          // 512
#define MM    (BB * SS)           // 4096
#define SCALE 0.125f              // 1/sqrt(64)
#define NEG_INF -1e30f

typedef _Float16 v16h __attribute__((ext_vector_type(16)));
typedef float    v8f  __attribute__((ext_vector_type(8)));
typedef unsigned int u32x4 __attribute__((ext_vector_type(4)));
typedef int          i32x8 __attribute__((ext_vector_type(8)));
typedef int          i32x4 __attribute__((ext_vector_type(4)));

union FragU    { uint4 u[2]; v16h v; };        // 32B fragment bit-cast
union HalfVecU { uint4 u; _Float16 h[8]; };    // 16B packed-f16 bit-cast

#if __has_builtin(__builtin_amdgcn_tensor_load_to_lds) && \
    __has_builtin(__builtin_amdgcn_s_wait_tensorcnt)
#define USE_TDM 1
#else
#define USE_TDM 0
#endif

__device__ __forceinline__ v8f wmma_f16(v16h a, v16h b, v8f c) {
  // D = A(16x32 f16) * B(32x16 f16) + C(16x16 f32)
  return __builtin_amdgcn_wmma_f32_16x16x32_f16(false, a, false, b, (short)0, c,
                                                false, false);
}

// ---- Vectorized fragment loaders (wave32 layouts, CDNA5 ISA 7.12.2) ----

// A-fragment 16x32 (MxK), row-major source, leading dim ld (f16 elements).
__device__ __forceinline__ v16h load_a_frag_f16(const _Float16* p, int ld) {
  const int lane = (int)(threadIdx.x & 31u);
  const int m  = lane & 15;
  const int kh = (lane >> 4) << 3;  // 0 or 8
  const _Float16* r = p + (size_t)m * (size_t)ld + kh;
  FragU f;
  f.u[0] = *reinterpret_cast<const uint4*>(r);
  f.u[1] = *reinterpret_cast<const uint4*>(r + 16);
  return f.v;
}

// B-fragment 32x16 (KxN), "weight-style" source: row n holds the K-contiguous
// column n of B (two 16-byte chunks per lane: k = kb..kb+7 and kb+16..kb+23).
__device__ __forceinline__ v16h load_b_frag_f16(const _Float16* p, int ld) {
  const int lane = (int)(threadIdx.x & 31u);
  const int n  = lane & 15;
  const int kb = (lane >> 4) << 3;  // 0 or 8
  const _Float16* r = p + (size_t)n * (size_t)ld + kb;
  FragU f;
  f.u[0] = *reinterpret_cast<const uint4*>(r);
  f.u[1] = *reinterpret_cast<const uint4*>(r + 16);
  return f.v;
}

#if USE_TDM
// ---- Tensor Data Mover: 2-D tile (h rows x w f16 elems) global -> LDS ------
// D# per CDNA5 ISA: group0 = {count, lds_addr, global_addr,
// type=2}; group1 = {data_size=2B, tensor dims == tile dims (in-bounds),
// tile dims, dim0 stride}. workgroup_mask=0 (not a cluster dispatch).
__device__ __forceinline__ void tdm_load_2d(const void* gsrc, void* ldst,
                                            uint32_t w, uint32_t h,
                                            uint32_t stride_elems) {
  const uint64_t ga = (uint64_t)(uintptr_t)gsrc;
  const uint32_t lo = (uint32_t)(uintptr_t)ldst;  // addr[31:0] == LDS offset
  u32x4 g0;
  g0[0] = 1u;                                               // count=1, user D#
  g0[1] = lo;                                               // lds_addr (bytes)
  g0[2] = (uint32_t)ga;                                     // global_addr lo32
  g0[3] = (uint32_t)((ga >> 32) & 0x01FFFFFFu) | (2u << 30);// addr[56:32]|type
  i32x8 g1;
  g1[0] = (int)(1u << 16);                      // data_size=1 -> 2 bytes
  g1[1] = (int)((w & 0xFFFFu) << 16);           // tensor_dim0[15:0]
  g1[2] = (int)((w >> 16) | ((h & 0xFFFFu) << 16)); // td0 hi | td1 lo
  g1[3] = (int)((h >> 16) | ((w & 0xFFFFu) << 16)); // td1 hi | tile_dim0
  g1[4] = (int)(h & 0xFFFFu);                   // tile_dim1 | tile_dim2=0
  g1[5] = (int)stride_elems;                    // tensor_dim0_stride lo32
  g1[6] = 0;
  g1[7] = 0;
  i32x4 z4 = {0, 0, 0, 0};
#if defined(__clang_major__) && (__clang_major__ >= 23)
  i32x8 z8 = {0, 0, 0, 0, 0, 0, 0, 0};
  __builtin_amdgcn_tensor_load_to_lds(g0, g1, z4, z4, z8, 0);
#else
  __builtin_amdgcn_tensor_load_to_lds(g0, g1, z4, z4, 0);
#endif
}
#endif  // USE_TDM

// -------------------- f32 -> f16 elementwise conversion --------------------
__global__ __launch_bounds__(256) void cvt_f32_f16(
    const float* __restrict__ src, _Float16* __restrict__ dst, int n) {
  const int i = ((int)blockIdx.x * 256 + (int)threadIdx.x) * 8;
  if (i >= n) return;
  const float4 a = reinterpret_cast<const float4*>(src + i)[0];
  const float4 b = reinterpret_cast<const float4*>(src + i)[1];
  HalfVecU hv;
  hv.h[0] = (_Float16)a.x; hv.h[1] = (_Float16)a.y;
  hv.h[2] = (_Float16)a.z; hv.h[3] = (_Float16)a.w;
  hv.h[4] = (_Float16)b.x; hv.h[5] = (_Float16)b.y;
  hv.h[6] = (_Float16)b.z; hv.h[7] = (_Float16)b.w;
  *reinterpret_cast<uint4*>(dst + i) = hv.u;
}

// -------------------- Projection GEMM: C = A * W^T + bias ------------------
// A: [M,K] f16, W: [N,K] f32 (converted to f16 while staged in LDS),
// C: [M,N] (OT) or, if TRANS, f16 [N,M] (for V^T).
// Block = 8 waves; block tile 256(M) x 64(N); wave tile 32x64.
template <typename OT, bool TRANS>
__global__ __launch_bounds__(256) void gemm_xwT_wmma(
    const _Float16* __restrict__ A, const float* __restrict__ W,
    const float* __restrict__ bias, OT* __restrict__ C,
    int M, int N, int K) {
  __shared__ _Float16 sB[64 * 32];

  const int tid  = (int)threadIdx.x;
  const int wave = tid >> 5;
  const int lane = tid & 31;
  const int m0 = (int)blockIdx.y * 256 + wave * 32;
  const int n0 = (int)blockIdx.x * 64;

  const int sn = tid >> 2;
  const int sk = (tid & 3) << 3;
  const float* wrow = W + (size_t)(n0 + sn) * (size_t)K + sk;

  v8f acc0[4] = {v8f{}, v8f{}, v8f{}, v8f{}};
  v8f acc1[4] = {v8f{}, v8f{}, v8f{}, v8f{}};

  for (int k0 = 0; k0 < K; k0 += 32) {
    __syncthreads();
    {
      const float* src = wrow + k0;
      HalfVecU hv;
#pragma unroll
      for (int i = 0; i < 8; ++i) hv.h[i] = (_Float16)src[i];
      *reinterpret_cast<uint4*>(&sB[sn * 32 + sk]) = hv.u;
      if (k0 + 32 < K) __builtin_prefetch(src + 32, 0, 0);  // global_prefetch
    }
    __syncthreads();

    const v16h a0 = load_a_frag_f16(A + (size_t)m0 * K + k0, K);
    const v16h a1 = load_a_frag_f16(A + (size_t)(m0 + 16) * K + k0, K);
#pragma unroll
    for (int t = 0; t < 4; ++t) {
      const v16h b = load_b_frag_f16(&sB[t * 16 * 32], 32);  // ds_load_b128
      acc0[t] = wmma_f16(a0, b, acc0[t]);
      acc1[t] = wmma_f16(a1, b, acc1[t]);
    }
  }

  const int mh = (lane >> 4) << 3;
  const int nc = lane & 15;
#pragma unroll
  for (int half = 0; half < 2; ++half) {
    const int mbase = m0 + half * 16 + mh;
    const v8f* accp = half ? acc1 : acc0;
#pragma unroll
    for (int t = 0; t < 4; ++t) {
      const int n = n0 + t * 16 + nc;
      const float bv = bias[n];
      if constexpr (TRANS) {
        HalfVecU hv;
#pragma unroll
        for (int r = 0; r < 8; ++r) hv.h[r] = (_Float16)(accp[t][r] + bv);
        _Float16* ct = (_Float16*)C;
        *reinterpret_cast<uint4*>(&ct[(size_t)n * (size_t)M + mbase]) = hv.u;
      } else {
#pragma unroll
        for (int r = 0; r < 8; ++r)
          C[(size_t)(mbase + r) * (size_t)N + n] = (OT)(accp[t][r] + bv);
      }
    }
  }
}

// -------------------- Sliding-window causal ALiBi attention ----------------
// Q: [M, H*DH] f16, Kws: [M, HKV*DH] f16, Vt: [HKV*DH, M] f16 (transposed),
// O: [M, H*DH] f16.
// Block = 8 waves, one (b,h), 128 consecutive queries (wave w: 16-query tile).
// 64-key blocks; K [64x64] and V [64(d)x64(key)] tiles staged in LDS by the
// Tensor Data Mover, double-buffered, shared by all 8 waves.
// grid = (S/128, H, B)
__global__ __launch_bounds__(256) void attn_swa_alibi_wmma(
    const _Float16* __restrict__ Q, const _Float16* __restrict__ Kws,
    const _Float16* __restrict__ Vt, const float* __restrict__ slopes,
    _Float16* __restrict__ O) {
  // LDS: K 2x[64x64] | V 2x[64x64] | per-wave P/O bounce 8x[16x64]  = 48KB
  __shared__ _Float16 smem[2 * 4096 + 2 * 4096 + 8 * 1024];

  const int wave = (int)(threadIdx.x >> 5);
  const int lane = (int)(threadIdx.x & 31u);
  const int b = (int)blockIdx.z;
  const int h = (int)blockIdx.y;
  const int qb = (int)blockIdx.x * 128;  // block query base
  const int q0 = qb + wave * 16;         // this wave's query tile
  const int hkv = h / GROUP;
  const float slope = slopes[h];

  const _Float16* Qb = Q   + ((size_t)(b * SS + q0) * DD) + h * DH;
  const _Float16* Kb = Kws + ((size_t)(b * SS) * KV_D) + hkv * DH;
  const _Float16* Vb = Vt  + (size_t)(hkv * DH) * MM + b * SS;

  _Float16* pb = smem + 16384 + wave * 1024;

  const v16h qa0 = load_a_frag_f16(Qb, DD);
  const v16h qa1 = load_a_frag_f16(Qb + 32, DD);

  v8f   o[4] = {v8f{}, v8f{}, v8f{}, v8f{}};
  float mrow[8], lrow[8];
#pragma unroll
  for (int r = 0; r < 8; ++r) { mrow[r] = NEG_INF; lrow[r] = 0.0f; }

  const int mh = (lane >> 4) << 3;
  const int nc = lane & 15;

  // Loop-invariant per-lane relative-position terms:
  //   d(q,key) = (q0 - j0 - kt*16) + base[r];  bias = -slope * d
  int   base[8];
  float fb[8];
#pragma unroll
  for (int r = 0; r < 8; ++r) {
    base[r] = mh + r - nc;
    fb[r]   = -slope * (float)base[r];
  }

  // Union key range for the whole block (all 8 waves' windows), 64-aligned.
  int jstart = qb - (WIN - 1);
  if (jstart < 0) jstart = 0;
  jstart &= ~63;
  const int nblk = (qb + 127 - jstart) / 64 + 1;

#if USE_TDM
  // Prologue: prefetch tiles 0 and 1 (2 TDM loads per tile: K and V)
  if (wave == 0) {
    tdm_load_2d(Kb + (size_t)jstart * KV_D, smem, 64, 64, KV_D);
    tdm_load_2d(Vb + jstart, smem + 8192, 64, 64, MM);
    if (nblk > 1) {
      tdm_load_2d(Kb + (size_t)(jstart + 64) * KV_D, smem + 4096, 64, 64, KV_D);
      tdm_load_2d(Vb + (jstart + 64), smem + 12288, 64, 64, MM);
    }
  }
#endif

  for (int it = 0; it < nblk; ++it) {
    const int j0 = jstart + it * 64;
    _Float16* sK = smem + (it & 1) * 4096;
    _Float16* sV = smem + 8192 + (it & 1) * 4096;

#if USE_TDM
    if (wave == 0) {
      if (it + 1 < nblk) __builtin_amdgcn_s_wait_tensorcnt(2);
      else               __builtin_amdgcn_s_wait_tensorcnt(0);
    }
    __syncthreads();  // publish TDM-written tile to all waves
#else
    // Fallback: cooperative single-buffered staging with vector loads
    __syncthreads();
    {
      const int tid = (int)threadIdx.x;
      const int rr = tid >> 2, cc = (tid & 3) << 4;  // 64 rows x 64 elems
      const uint4* gk =
          reinterpret_cast<const uint4*>(Kb + (size_t)(j0 + rr) * KV_D + cc);
      uint4* lk = reinterpret_cast<uint4*>(&sK[rr * 64 + cc]);
      lk[0] = gk[0]; lk[1] = gk[1];
      const uint4* gv =
          reinterpret_cast<const uint4*>(Vb + (size_t)rr * MM + j0 + cc);
      uint4* lv = reinterpret_cast<uint4*>(&sV[rr * 64 + cc]);
      lv[0] = gv[0]; lv[1] = gv[1];
    }
    __syncthreads();
#endif

    // Wave-uniform activity test: does this key block touch our window?
    const bool active = (j0 <= q0 + 15) && (j0 + 63 >= q0 - (WIN - 1));
    if (active) {
      // ---- scores S[16q, 64k] as four 16x16 f32 tiles (K from LDS) ----
      v8f s[4];
#pragma unroll
      for (int kt = 0; kt < 4; ++kt) {
        const _Float16* kp = sK + (kt * 16) * 64;
        v8f c = v8f{};
        c = wmma_f16(qa0, load_b_frag_f16(kp, 64), c);
        c = wmma_f16(qa1, load_b_frag_f16(kp + 32, 64), c);
        s[kt] = c;
      }

      // ---- scale + ALiBi + causal/window mask (no cvt/mul in loop) ----
#pragma unroll
      for (int kt = 0; kt < 4; ++kt) {
        const int   sdk = (q0 - j0) - kt * 16;   // wave-uniform
        const float sbf = -slope * (float)sdk;   // wave-uniform
#pragma unroll
        for (int r = 0; r < 8; ++r) {
          const int d = sdk + base[r];           // q - key
          const bool valid = (unsigned)d < (unsigned)WIN;
          const float v = s[kt][r] * SCALE + (sbf + fb[r]);
          s[kt][r] = valid ? v : NEG_INF;
        }
      }

      // ---- online softmax (row reductions across 16-lane halves) ----
      float pm[8];
#pragma unroll
      for (int r = 0; r < 8; ++r) {
        float v = fmaxf(fmaxf(s[0][r], s[1][r]), fmaxf(s[2][r], s[3][r]));
#pragma unroll
        for (int off = 1; off <= 8; off <<= 1)
          v = fmaxf(v, __shfl_xor(v, off, 32));
        pm[r] = v;
      }

      float cfac[8];
#pragma unroll
      for (int r = 0; r < 8; ++r) {
        const float nm = fmaxf(mrow[r], pm[r]);
        cfac[r] = (mrow[r] <= NEG_INF * 0.5f) ? 0.0f : __expf(mrow[r] - nm);
        mrow[r] = nm;
      }

      float rs[8];
#pragma unroll
      for (int r = 0; r < 8; ++r) rs[r] = 0.0f;
#pragma unroll
      for (int kt = 0; kt < 4; ++kt) {       // p computed in-place over s
#pragma unroll
        for (int r = 0; r < 8; ++r) {
          const float sv = s[kt][r];
          const float pv = (sv <= NEG_INF * 0.5f) ? 0.0f : __expf(sv - mrow[r]);
          s[kt][r] = pv;
          rs[r] += pv;
        }
      }
#pragma unroll
      for (int r = 0; r < 8; ++r) {
        float v = rs[r];
#pragma unroll
        for (int off = 1; off <= 8; off <<= 1) v += __shfl_xor(v, off, 32);
        lrow[r] = lrow[r] * cfac[r] + v;
      }

#pragma unroll
      for (int t = 0; t < 4; ++t)
#pragma unroll
        for (int r = 0; r < 8; ++r) o[t][r] *= cfac[r];

      // ---- P: C-layout -> A-fragment layout via wave-private LDS ----
#pragma unroll
      for (int kt = 0; kt < 4; ++kt)
#pragma unroll
        for (int r = 0; r < 8; ++r)
          pb[(mh + r) * 64 + kt * 16 + nc] = (_Float16)s[kt][r];
      const v16h pa0 = load_a_frag_f16(pb, 64);       // keys j0..j0+31
      const v16h pa1 = load_a_frag_f16(pb + 32, 64);  // keys j0+32..j0+63

      // ---- O += P(16x64) * V(64x64); V tile in LDS is d-major ----
#pragma unroll
      for (int t = 0; t < 4; ++t) {
        const _Float16* vp = sV + (t * 16) * 64;
        o[t] = wmma_f16(pa0, load_b_frag_f16(vp, 64), o[t]);
        o[t] = wmma_f16(pa1, load_b_frag_f16(vp + 32, 64), o[t]);
      }
    }

    __syncthreads();  // everyone done with buffer (it&1) before reuse

#if USE_TDM
    if (wave == 0 && it + 2 < nblk) {
      const int j2 = jstart + (it + 2) * 64;
      _Float16* nK = smem + (it & 1) * 4096;          // reuse the freed buffer
      _Float16* nV = smem + 8192 + (it & 1) * 4096;
      tdm_load_2d(Kb + (size_t)j2 * KV_D, nK, 64, 64, KV_D);
      tdm_load_2d(Vb + j2, nV, 32 + 32, 64, MM);
    }
#endif
  }

  // ---- normalize, bounce through LDS, and store context coalesced ----
#pragma unroll
  for (int r = 0; r < 8; ++r) lrow[r] = 1.0f / lrow[r];
#pragma unroll
  for (int t = 0; t < 4; ++t)
#pragma unroll
    for (int r = 0; r < 8; ++r)
      pb[(mh + r) * 64 + t * 16 + nc] = (_Float16)(o[t][r] * lrow[r]);
  {
    // lane pair covers one query row: 32 contiguous f16 -> 4 b128 stores
    const int row = lane >> 1;
    const int col = (lane & 1) * 32;
    const uint4* src = reinterpret_cast<const uint4*>(&pb[row * 64 + col]);
    uint4* dst = reinterpret_cast<uint4*>(
        O + (size_t)(b * SS + q0 + row) * DD + h * DH + col);
#pragma unroll
    for (int i = 0; i < 4; ++i) dst[i] = src[i];
  }
}

// --------------------------------------------------------------------------

extern "C" void kernel_launch(void* const* d_in, const int* in_sizes, int n_in,
                              void* d_out, int out_size, void* d_ws,
                              size_t ws_size, hipStream_t stream) {
  (void)in_sizes; (void)n_in; (void)out_size; (void)ws_size;
  const float* x      = (const float*)d_in[0];
  const float* Wq     = (const float*)d_in[1];
  const float* bq     = (const float*)d_in[2];
  const float* Wk     = (const float*)d_in[3];
  const float* bk     = (const float*)d_in[4];
  const float* Wv     = (const float*)d_in[5];
  const float* bv     = (const float*)d_in[6];
  const float* Wo     = (const float*)d_in[7];
  const float* bo     = (const float*)d_in[8];
  const float* slopes = (const float*)d_in[9];
  float* out = (float*)d_out;

  // Workspace (f16): xh | Q | K | V^T | attn  -> ~59 MB, L2-resident.
  _Float16* xh  = (_Float16*)d_ws;
  _Float16* qws = xh  + (size_t)MM * DD;
  _Float16* kws = qws + (size_t)MM * DD;
  _Float16* vt  = kws + (size_t)MM * KV_D;
  _Float16* aws = vt  + (size_t)KV_D * MM;

  cvt_f32_f16<<<dim3((MM * DD) / 8 / 256), 256, 0, stream>>>(x, xh, MM * DD);

  gemm_xwT_wmma<_Float16, false>
      <<<dim3(DD / 64, MM / 256), 256, 0, stream>>>(xh, Wq, bq, qws, MM, DD, DD);
  gemm_xwT_wmma<_Float16, false>
      <<<dim3(KV_D / 64, MM / 256), 256, 0, stream>>>(xh, Wk, bk, kws, MM, KV_D, DD);
  gemm_xwT_wmma<_Float16, true>
      <<<dim3(KV_D / 64, MM / 256), 256, 0, stream>>>(xh, Wv, bv, vt, MM, KV_D, DD);

  attn_swa_alibi_wmma<<<dim3(SS / 128, HH, BB), 256, 0, stream>>>(
      qws, kws, vt, slopes, aws);

  gemm_xwT_wmma<float, false>
      <<<dim3(DD / 64, MM / 256), 256, 0, stream>>>(aws, Wo, bo, out, MM, DD, DD);
}
